// SRLEmbeddings_18451179503677
// MI455X (gfx1250) — matrix operand order repaired and checked by reading, not compiled, not run on hardware
//
#include <hip/hip_runtime.h>
#include <hip/hip_bf16.h>

// CDNA5 / gfx1250 WMMA types
typedef __attribute__((ext_vector_type(16))) _Float16 v16h;
typedef __attribute__((ext_vector_type(8)))  float    v8f;

#define L_        256
#define D_        768
#define P_        16
#define T_        4
#define PAD_ID    1
#define CHUNK     32
#define NCHUNK    8           // L_ / CHUNK
#define NTHREADS  384         // 12 wave32
#define NWAVES    12
#define TPW       4           // D tiles (16 wide) per wave: 48 tiles / 12 waves

// Output layout in d_out (flat, reference tuple order):
//   sent  [B,S,D]      393216 floats @ 0
//   pred  [B,S,P,D]  6291456 floats @ 393216
//   arg0  [B,S,P,D]  6291456 floats @ 6684672
//   arg1  [B,S,P,D]  6291456 floats @ 12976128

__global__ __launch_bounds__(NTHREADS, 1)
void srl_wmma_kernel(const int* __restrict__ g_sid,    // [B,S,L]
                     const int* __restrict__ g_attn,   // [B,S,L]
                     const int* __restrict__ g_pid,    // [B,S,P,T]
                     const int* __restrict__ g_a0,
                     const int* __restrict__ g_a1,
                     const float* __restrict__ g_emb,  // [B,S,L,D] f32
                     float* __restrict__ g_out)
{
    __shared__ int      sh_sid[L_];
    __shared__ float    sh_attn[L_];
    __shared__ int      sh_ids[3][P_][T_];
    __shared__ unsigned sh_den[3][P_];
    // A fragments (counts) for current K-chunk, packed per WMMA 16-bit A layout:
    // sh_cntA[arg][lane][v] = u32 of two f16 counts (k0, k0+1), 32B/lane contiguous
    __shared__ __align__(32) unsigned sh_cntA[3][32][8];
    // B fragments (f16 embeddings) for current K-chunk:
    // sh_EB[d][pair] = u32 of (E[k=2*pair][d], E[k=2*pair+1][d]); lane reads 8 u32 @ [d][8*half]
    __shared__ __align__(64) unsigned sh_EB[D_][16];
    __shared__ float    sh_tokcnt;

    const int tid  = threadIdx.x;
    const int lane = tid & 31;
    const int wave = tid >> 5;
    const int bs   = blockIdx.x;              // 0..511

    const int*   sid_b  = g_sid  + bs * L_;
    const int*   attn_b = g_attn + bs * L_;
    const float* emb_b  = g_emb  + (size_t)bs * L_ * D_;

    // ---- phase 0: stage metadata into LDS -------------------------------
    if (tid < L_) {
        sh_sid[tid]  = sid_b[tid];
        sh_attn[tid] = (float)attn_b[tid];
    }
    if (tid >= L_ && tid < L_ + 128) {        // 128 threads load 3*64 span ids (first 2/3)
        const int i = tid - L_;
        const int* src = (i < 64) ? g_pid : g_a0;
        ((int*)sh_ids)[i] = src[bs * (P_ * T_) + (i & 63)];
    }
    if (tid < 64) {                            // last 64 span ids
        ((int*)sh_ids)[128 + tid] = g_a1[bs * (P_ * T_) + tid];
    }
    if (tid >= 320 && tid < 368) ((unsigned*)sh_den)[tid - 320] = 0u;
    __syncthreads();

    if (tid == 0) {
        float s = 0.f;
        for (int l = 0; l < L_; ++l) s += sh_attn[l];
        sh_tokcnt = (s < 1.f) ? 1.f : s;      // clip(sum, 1, inf)
    }
    // denominators: 4 threads per (arg,p), 64 tokens each, shared atomic combine
    if (tid < 192) {
        const int arg = tid >> 6;
        const int r   = tid & 63;
        const int p   = r >> 2;
        const int q   = r & 3;
        const int id0 = sh_ids[arg][p][0], id1 = sh_ids[arg][p][1];
        const int id2 = sh_ids[arg][p][2], id3 = sh_ids[arg][p][3];
        unsigned cnt = 0;
        for (int l = q * 64; l < q * 64 + 64; ++l) {
            if (sh_attn[l] != 0.f) {
                const int sv = sh_sid[l];
                cnt += (unsigned)((id0 != PAD_ID) & (id0 == sv));
                cnt += (unsigned)((id1 != PAD_ID) & (id1 == sv));
                cnt += (unsigned)((id2 != PAD_ID) & (id2 == sv));
                cnt += (unsigned)((id3 != PAD_ID) & (id3 == sv));
            }
        }
        atomicAdd(&((unsigned*)sh_den)[arg * P_ + p], cnt);
    }

    // ---- persistent accumulators: 4 tiles x 3 span types x 16x16 f32 ----
    const v8f vzero = {};
    v8f acc[TPW][3];
#pragma unroll
    for (int t = 0; t < TPW; ++t)
#pragma unroll
        for (int a = 0; a < 3; ++a) acc[t][a] = vzero;

    float sa0 = 0.f, sa1 = 0.f;               // sentence-sum partials (2 columns/thread)
    _Float16* ebh = (_Float16*)&sh_EB[0][0];  // f16 view: index d*CHUNK + k

    // ---- main loop over K (= L) in chunks of 32 -------------------------
    for (int c = 0; c < NCHUNK; ++c) {
        // stage E chunk: coalesced f32 reads, f16 packed into B-frag layout;
        // column d = tid and d = tid+384 owned by exactly one thread -> no atomics
#pragma unroll 4
        for (int li = 0; li < CHUNK; ++li) {
            const int l = c * CHUNK + li;
            const float* row = emb_b + (size_t)l * D_;
            const float m  = sh_attn[l];
            const float v0 = row[tid];
            const float v1 = row[tid + NTHREADS];
            sa0 += m * v0;  sa1 += m * v1;
            ebh[(size_t)(tid           ) * CHUNK + li] = (_Float16)v0;
            ebh[(size_t)(tid + NTHREADS) * CHUNK + li] = (_Float16)v1;
        }
        // build count A-fragments for this chunk, directly in WMMA layout
#pragma unroll
        for (int s3 = 0; s3 < 2; ++s3) {
            const int slot = tid + s3 * NTHREADS;        // 0..767 = 3*32*8 slots
            const int arg  = slot >> 8;
            const int rr   = slot & 255;
            const int ln   = rr >> 3;                    // target lane
            const int v    = rr & 7;                     // VGPR pair index
            const int hf   = ln >> 4;
            const int mm   = ln & 15;                    // matrix row M (= predicate p)
            const int k0   = 2 * v + 16 * hf;  // contiguous k = 16*half + s, same order as B
            const int l0   = c * CHUNK + k0;
            const int id0 = sh_ids[arg][mm][0], id1 = sh_ids[arg][mm][1];
            const int id2 = sh_ids[arg][mm][2], id3 = sh_ids[arg][mm][3];
            int c0 = 0, c1 = 0;
            if (sh_attn[l0] != 0.f) {
                const int sv = sh_sid[l0];
                c0 = ((id0 != PAD_ID) & (id0 == sv)) + ((id1 != PAD_ID) & (id1 == sv))
                   + ((id2 != PAD_ID) & (id2 == sv)) + ((id3 != PAD_ID) & (id3 == sv));
            }
            if (sh_attn[l0 + 1] != 0.f) {
                const int sv = sh_sid[l0 + 1];
                c1 = ((id0 != PAD_ID) & (id0 == sv)) + ((id1 != PAD_ID) & (id1 == sv))
                   + ((id2 != PAD_ID) & (id2 == sv)) + ((id3 != PAD_ID) & (id3 == sv));
            }
            _Float16* ca = (_Float16*)&sh_cntA[arg][ln][0];
            ca[2 * v]     = (_Float16)(float)c0;
            ca[2 * v + 1] = (_Float16)(float)c1;
        }
        __syncthreads();

        // WMMA: A frags shared across tiles, B frag shared across the 3 span types
        const v16h afr0 = *(const v16h*)&sh_cntA[0][lane][0];
        const v16h afr1 = *(const v16h*)&sh_cntA[1][lane][0];
        const v16h afr2 = *(const v16h*)&sh_cntA[2][lane][0];
#pragma unroll
        for (int tl = 0; tl < TPW; ++tl) {
            const int d0 = wave * (TPW * 16) + tl * 16;
            const v16h bfr = *(const v16h*)&sh_EB[d0 + (lane & 15)][(lane >> 4) * 8];
            acc[tl][0] = __builtin_amdgcn_wmma_f32_16x16x32_f16(
                false, afr0, false, bfr, (short)0, acc[tl][0], false, false);
            acc[tl][1] = __builtin_amdgcn_wmma_f32_16x16x32_f16(
                false, afr1, false, bfr, (short)0, acc[tl][1], false, false);
            acc[tl][2] = __builtin_amdgcn_wmma_f32_16x16x32_f16(
                false, afr2, false, bfr, (short)0, acc[tl][2], false, false);
        }
        __syncthreads();   // before next chunk overwrites LDS
    }

    // ---- epilogue: sentence mean ---------------------------------------
    const float inv = 1.0f / sh_tokcnt;
    float* out_sent = g_out + (size_t)bs * D_;
    out_sent[tid]            = sa0 * inv;
    out_sent[tid + NTHREADS] = sa1 * inv;

    // ---- epilogue: span embeddings (C layout: M = r + 8*half, N = lane&15)
    const size_t ARG_OFF0 = 393216u;     // pred
    const size_t ARG_OFF1 = 6684672u;    // arg0
    const size_t ARG_OFF2 = 12976128u;   // arg1
    float* outA[3];
    outA[0] = g_out + ARG_OFF0 + (size_t)bs * (P_ * D_);
    outA[1] = g_out + ARG_OFF1 + (size_t)bs * (P_ * D_);
    outA[2] = g_out + ARG_OFF2 + (size_t)bs * (P_ * D_);

#pragma unroll
    for (int tl = 0; tl < TPW; ++tl) {
        const int dcol = wave * (TPW * 16) + tl * 16 + (lane & 15);
#pragma unroll
        for (int a = 0; a < 3; ++a) {
#pragma unroll
            for (int r = 0; r < 8; ++r) {
                const int p = r + 8 * (lane >> 4);
                const float den = (float)sh_den[a][p];
                const float val = (den > 0.f) ? (acc[tl][a][r] / den) : 0.f;
                outA[a][(size_t)p * D_ + dcol] = val;
            }
        }
    }
}

extern "C" void kernel_launch(void* const* d_in, const int* in_sizes, int n_in,
                              void* d_out, int out_size, void* d_ws, size_t ws_size,
                              hipStream_t stream) {
    (void)in_sizes; (void)n_in; (void)out_size; (void)d_ws; (void)ws_size;
    const int*   sid  = (const int*)d_in[0];
    const int*   attn = (const int*)d_in[1];
    const int*   pid  = (const int*)d_in[2];
    const int*   a0   = (const int*)d_in[3];
    const int*   a1   = (const int*)d_in[4];
    const float* emb  = (const float*)d_in[5];
    float* out = (float*)d_out;
    // one workgroup per (b,s): 16*32 = 512 blocks, 384 threads = 12 wave32
    srl_wmma_kernel<<<512, NTHREADS, 0, stream>>>(sid, attn, pid, a0, a1, emb, out);
}
